// WindowCuttingPosition_62612033241207
// MI455X (gfx1250) — matrix operand order repaired and checked by reading, not compiled, not run on hardware
//
#include <hip/hip_runtime.h>
#include <hip/hip_bf16.h>

// ---------------------------------------------------------------------------
// Fused Swin window attention for MI455X (gfx1250, wave32, WMMA).
// One workgroup (8 wave32) per 7x7 window; QKV, attention, softmax, PV,
// out-proj, residual+LN all stay in LDS. bf16 WMMA 16x16x32 is the workhorse
// (K=32 == head_dim). Weights pre-transposed+converted to bf16 in d_ws so
// WMMA B-fragments are single contiguous 32B loads. The f32 input window is
// staged into LDS with GLOBAL_LOAD_ASYNC_TO_LDS_B128 (ASYNCcnt path).
// ---------------------------------------------------------------------------

#define NTOK 49
#define NPAD 64
#define CH 256
#define HEADS 8
#define HD 32
#define C3 768

typedef __attribute__((ext_vector_type(16))) __bf16 bf16x16;
typedef __attribute__((ext_vector_type(8)))  float  floatx8;

union FragU { bf16x16 v; unsigned u[8]; __bf16 h[16]; };

__device__ __forceinline__ floatx8 wmma_bf16(bf16x16 a, bf16x16 b, floatx8 c) {
  // D = A(16x32 bf16) * B(32x16 bf16) + C(16x16 f32)
  return __builtin_amdgcn_wmma_f32_16x16x32_bf16(false, a, false, b, (short)0, c,
                                                 false, false);
}

// 16-bit A-matrix fragment (16x32), row-major source buf[row][stride].
// Lane l: row = row0 + (l&15); lanes<16 hold K {0..15}, lanes>=16 hold
// K {16..31}; pairs pack per 32-bit VGPR, so vg0..3 and
// vg4..7 each merge into one ds_load_b128.
__device__ __forceinline__ bf16x16 load_a_frag(const __bf16* buf, int stride,
                                               int row0, int k0) {
  const int lane = threadIdx.x & 31;
  const int row  = row0 + (lane & 15);
  const int kb   = (lane < 16) ? 0 : 16;
  const __bf16* p = buf + row * stride + k0 + kb;
  FragU f;
#pragma unroll
  for (int vg = 0; vg < 8; ++vg) {
    const int kl = vg << 1;
    f.u[vg] = *(const unsigned*)(p + kl);
  }
  return f.v;
}

// 16-bit B-matrix fragment (32x16) from buf[k][stride] with k strided
// (used for V where the K dimension is the token index).
__device__ __forceinline__ bf16x16 load_b_frag_strided(const __bf16* buf, int stride,
                                                       int k0, int col0) {
  const int lane = threadIdx.x & 31;
  const int col  = col0 + (lane & 15);
  const int kb   = (lane < 16) ? 0 : 16;
  FragU f;
#pragma unroll
  for (int e = 0; e < 16; ++e) f.h[e] = buf[(k0 + kb + e) * stride + col];
  return f.v;
}

// ---------------------------------------------------------------------------
// Prep: transpose+convert weights to bf16, materialize relative-position bias.
// wqkvT[n][k] (768x256 bf16), woutT[n][k] (256x256 bf16), bias[h][q][k] f32.
// ---------------------------------------------------------------------------
__global__ void swin_prep_kernel(const float* __restrict__ w_qkv,
                                 const float* __restrict__ w_out,
                                 const float* __restrict__ bias_table,
                                 __bf16* __restrict__ wqkvT,
                                 __bf16* __restrict__ woutT,
                                 float* __restrict__ bias) {
  const int stride = gridDim.x * blockDim.x;
  const int t0 = blockIdx.x * blockDim.x + threadIdx.x;
  for (int i = t0; i < C3 * CH; i += stride) {
    const int n = i >> 8, k = i & 255;
    wqkvT[i] = (__bf16)w_qkv[k * C3 + n];
  }
  for (int i = t0; i < CH * CH; i += stride) {
    const int n = i >> 8, k = i & 255;
    woutT[i] = (__bf16)w_out[k * CH + n];
  }
  for (int i = t0; i < HEADS * NTOK * NTOK; i += stride) {
    const int h = i / (NTOK * NTOK);
    const int r = i % (NTOK * NTOK);
    const int q = r / NTOK, kk = r % NTOK;
    const int rel = (q / 7 - kk / 7 + 6) * 13 + (q % 7 - kk % 7 + 6);
    bias[i] = bias_table[rel * HEADS + h];
  }
}

// ---------------------------------------------------------------------------
// Main fused kernel: grid = 4096 windows, block = 256 threads (8 wave32).
// Dynamic LDS: [0,32K)     x tile bf16 (64x256), reused as attn-output tile
//              [32K,128K)  qkv bf16 (64x768), later aliased as f32 y (64x256)
//              [128K,192K) per-head probs bf16 (8 x 64x64)
//              [192K,241K) raw f32 window (49x256) for residual, async-staged
// ---------------------------------------------------------------------------
__global__ __launch_bounds__(256, 1)
void swin_window_attn_kernel(const float* __restrict__ x,
                             const float* __restrict__ b_qkv,
                             const float* __restrict__ b_out,
                             const float* __restrict__ ln_gamma,
                             const float* __restrict__ ln_beta,
                             const __bf16* __restrict__ wqkvT,
                             const __bf16* __restrict__ woutT,
                             const float* __restrict__ bias,
                             float* __restrict__ out) {
  extern __shared__ __align__(128) char smem[];
  __bf16* xo  = (__bf16*)smem;                     // 64x256 bf16 (x, then O)
  __bf16* qkv = (__bf16*)(smem + 32768);           // 64x768 bf16
  float*  yf  = (float*)(smem + 32768);            // 64x256 f32 (aliases qkv)
  __bf16* prb = (__bf16*)(smem + 32768 + 98304);   // 8 x 64x64 bf16
  float*  xf  = (float*)(smem + 32768 + 98304 + 65536);  // 49x256 f32

  const int w    = blockIdx.x;
  const int b    = w >> 10;
  const int wy   = (w >> 5) & 31;
  const int wx   = w & 31;
  const int tid  = threadIdx.x;
  const int lane = tid & 31;
  const int wave = tid >> 5;
  const int colh = lane & 15;                  // N within a 16-wide tile
  const int rb   = (lane < 16) ? 0 : 8;        // C-layout row offset
  const int kbB  = (lane < 16) ? 0 : 16;       // B-layout K offset

  // ---- phase 0a: async-stage raw f32 window into LDS (ASYNCcnt path) ----
  for (int i = tid; i < NTOK * 64; i += 256) {       // 49 rows x 64 float4
    const int row = i >> 6;
    const int c4  = (i & 63) << 2;
    const int gr  = wy * 7 + row / 7;
    const int gc  = wx * 7 + row % 7;
    const float* gp = x + (((size_t)b * 224 + gr) * 224 + gc) * CH + c4;
    const unsigned ldsoff = (unsigned)(uintptr_t)(xf + row * CH + c4);
    asm volatile("global_load_async_to_lds_b128 %0, %1, off"
                 :: "v"(ldsoff), "v"(gp) : "memory");
  }
  asm volatile("s_wait_asynccnt 0x0" ::: "memory");
  __syncthreads();

  // ---- phase 0b: convert to bf16 tile from LDS (rows 49..63 zero) ----
  for (int i = tid; i < NPAD * 64; i += 256) {
    const int row = i >> 6;
    const int c4  = (i & 63) << 2;
    union { uint2 q; __bf16 h[4]; } u;
    if (row < NTOK) {
      const float4 v = *(const float4*)(xf + row * CH + c4);
      u.h[0] = (__bf16)v.x; u.h[1] = (__bf16)v.y;
      u.h[2] = (__bf16)v.z; u.h[3] = (__bf16)v.w;
    } else {
      u.q = make_uint2(0u, 0u);
    }
    *(uint2*)(xo + row * CH + c4) = u.q;
  }
  __syncthreads();

  // ---- phase 1: QKV GEMM (64x256 @ 256x768), 6 N-tiles per wave ----
#pragma unroll 1
  for (int j = 0; j < 6; ++j) {
    const int tn   = wave * 6 + j;
    const int colg = tn * 16 + colh;
    floatx8 acc[4];
#pragma unroll
    for (int tm = 0; tm < 4; ++tm) acc[tm] = (floatx8){0,0,0,0,0,0,0,0};
#pragma unroll 1
    for (int kk = 0; kk < 8; ++kk) {
      const bf16x16 bw =
          *(const bf16x16*)(wqkvT + (size_t)colg * CH + kk * 32 + kbB);
#pragma unroll
      for (int tm = 0; tm < 4; ++tm) {
        const bf16x16 ax = load_a_frag(xo, CH, tm * 16, kk * 32);
        acc[tm] = wmma_bf16(ax, bw, acc[tm]);
      }
    }
    const float bb = b_qkv[colg];
#pragma unroll
    for (int tm = 0; tm < 4; ++tm)
#pragma unroll
      for (int v = 0; v < 8; ++v)
        qkv[(tm * 16 + v + rb) * C3 + colg] = (__bf16)(acc[tm][v] + bb);
  }
  __syncthreads();

  // ---- phase 2: per-head (wave==head) scores + softmax -> probs bf16 ----
  const int h = wave;
  const float* biasg = bias + h * NTOK * NTOK;
  __bf16* pb = prb + h * NPAD * NPAD;

#pragma unroll 1
  for (int tm = 0; tm < 4; ++tm) {
    const bf16x16 aq = load_a_frag(qkv, C3, tm * 16, h * HD);   // Q
    float s[4][8];
#pragma unroll
    for (int tn = 0; tn < 4; ++tn) {
      const int colg = tn * 16 + colh;
      // B = K^T: column = key token, contiguous in d (head channel)
      const bf16x16 bk =
          *(const bf16x16*)(qkv + (size_t)colg * C3 + CH + h * HD + kbB);
      floatx8 z = (floatx8){0,0,0,0,0,0,0,0};
      const floatx8 a = wmma_bf16(aq, bk, z);
#pragma unroll
      for (int v = 0; v < 8; ++v) {
        const int row = tm * 16 + v + rb;
        float sv = a[v] * 0.5f;                 // scale = (heads//2)^-0.5
        if (colg < NTOK) {
          if (row < NTOK) sv += biasg[row * NTOK + colg];
        } else {
          sv = -1e30f;                          // mask padded keys
        }
        s[tn][v] = sv;
      }
    }
    // softmax over the 64-wide row: 4 tiles in registers + 16-lane shuffle
#pragma unroll
    for (int v = 0; v < 8; ++v) {
      float m = fmaxf(fmaxf(s[0][v], s[1][v]), fmaxf(s[2][v], s[3][v]));
#pragma unroll
      for (int msk = 8; msk >= 1; msk >>= 1) m = fmaxf(m, __shfl_xor(m, msk, 32));
      const float e0 = __expf(s[0][v] - m);
      const float e1 = __expf(s[1][v] - m);
      const float e2 = __expf(s[2][v] - m);
      const float e3 = __expf(s[3][v] - m);
      float sum = e0 + e1 + e2 + e3;
#pragma unroll
      for (int msk = 8; msk >= 1; msk >>= 1) sum += __shfl_xor(sum, msk, 32);
      const float r = 1.0f / sum;
      const int row = tm * 16 + v + rb;
      pb[row * NPAD +  0 + colh] = (__bf16)(e0 * r);
      pb[row * NPAD + 16 + colh] = (__bf16)(e1 * r);
      pb[row * NPAD + 32 + colh] = (__bf16)(e2 * r);
      pb[row * NPAD + 48 + colh] = (__bf16)(e3 * r);
    }
  }

  // ---- phase 3: O_head = P(64x64) @ V(64x32), write bf16 into xo ----
#pragma unroll 1
  for (int tm = 0; tm < 4; ++tm) {
#pragma unroll
    for (int tn = 0; tn < 2; ++tn) {
      floatx8 acc = (floatx8){0,0,0,0,0,0,0,0};
#pragma unroll
      for (int ks = 0; ks < 2; ++ks) {
        const bf16x16 ap = load_a_frag(pb, NPAD, tm * 16, ks * 32);
        const bf16x16 bv =
            load_b_frag_strided(qkv, C3, ks * 32, 2 * CH + h * HD + tn * 16);
        acc = wmma_bf16(ap, bv, acc);
      }
      const int cg = h * HD + tn * 16 + colh;
#pragma unroll
      for (int v = 0; v < 8; ++v)
        xo[(tm * 16 + v + rb) * CH + cg] = (__bf16)acc[v];
    }
  }
  __syncthreads();

  // ---- phase 4: out-proj (64x256 @ 256x256) + bias + residual -> yf ----
#pragma unroll 1
  for (int t2 = 0; t2 < 2; ++t2) {
    const int tn   = wave * 2 + t2;
    const int colg = tn * 16 + colh;
    floatx8 acc[4];
#pragma unroll
    for (int tm = 0; tm < 4; ++tm) acc[tm] = (floatx8){0,0,0,0,0,0,0,0};
#pragma unroll 1
    for (int kk = 0; kk < 8; ++kk) {
      const bf16x16 bw =
          *(const bf16x16*)(woutT + (size_t)colg * CH + kk * 32 + kbB);
#pragma unroll
      for (int tm = 0; tm < 4; ++tm) {
        const bf16x16 ao = load_a_frag(xo, CH, tm * 16, kk * 32);
        acc[tm] = wmma_bf16(ao, bw, acc[tm]);
      }
    }
    const float bo = b_out[colg];
#pragma unroll
    for (int tm = 0; tm < 4; ++tm)
#pragma unroll
      for (int v = 0; v < 8; ++v) {
        const int row = tm * 16 + v + rb;
        float y = acc[tm][v] + bo;
        if (row < NTOK) y += xf[row * CH + colg];   // residual from LDS
        yf[row * CH + colg] = y;
      }
  }
  __syncthreads();

  // ---- phase 5: LayerNorm, one wave per token, 8 channels per lane ----
#pragma unroll 1
  for (int t = wave; t < NTOK; t += HEADS) {
    const float* yr = yf + t * CH;
    const float4 v0 = *(const float4*)(yr + lane * 8);
    const float4 v1 = *(const float4*)(yr + lane * 8 + 4);
    float arr[8] = {v0.x, v0.y, v0.z, v0.w, v1.x, v1.y, v1.z, v1.w};
    float sum = 0.f, sq = 0.f;
#pragma unroll
    for (int jj = 0; jj < 8; ++jj) { sum += arr[jj]; sq += arr[jj] * arr[jj]; }
#pragma unroll
    for (int msk = 16; msk >= 1; msk >>= 1) {
      sum += __shfl_xor(sum, msk, 32);
      sq  += __shfl_xor(sq,  msk, 32);
    }
    const float mean = sum * (1.0f / CH);
    const float var  = sq * (1.0f / CH) - mean * mean;
    const float rstd = rsqrtf(var + 1e-5f);
    const size_t ob = ((size_t)w * NTOK + t) * CH + lane * 8;
#pragma unroll
    for (int jj = 0; jj < 8; ++jj) {
      const float g  = ln_gamma[lane * 8 + jj];
      const float bt = ln_beta[lane * 8 + jj];
      out[ob + jj] = (arr[jj] - mean) * rstd * g + bt;
    }
  }
}

extern "C" void kernel_launch(void* const* d_in, const int* in_sizes, int n_in,
                              void* d_out, int out_size, void* d_ws, size_t ws_size,
                              hipStream_t stream) {
  (void)in_sizes; (void)n_in; (void)out_size; (void)ws_size;
  const float* x     = (const float*)d_in[0];
  const float* w_qkv = (const float*)d_in[1];
  const float* b_qkv = (const float*)d_in[2];
  const float* btab  = (const float*)d_in[3];
  const float* w_out = (const float*)d_in[4];
  const float* b_out = (const float*)d_in[5];
  const float* gamma = (const float*)d_in[6];
  const float* beta  = (const float*)d_in[7];
  float* out = (float*)d_out;

  // Workspace carve: wqkvT (768x256 bf16), woutT (256x256 bf16), bias f32.
  __bf16* wqkvT = (__bf16*)d_ws;
  __bf16* woutT = (__bf16*)((char*)d_ws + (size_t)C3 * CH * 2);
  float*  biasb = (float*)((char*)d_ws + (size_t)C3 * CH * 2 + (size_t)CH * CH * 2);

  swin_prep_kernel<<<512, 256, 0, stream>>>(w_qkv, w_out, btab, wqkvT, woutT, biasb);

  // 32K xo + 96K qkv/yf + 64K probs + 49K xf = 241K of the 320 KB WGP LDS
  const size_t smem = 32768 + 98304 + 65536 + (size_t)NTOK * CH * 4;
  swin_window_attn_kernel<<<4096, 256, smem, stream>>>(
      x, b_qkv, b_out, gamma, beta, wqkvT, woutT, biasb, out);
}
